// APPNP_72035191489001
// MI455X (gfx1250) — matrix operand order repaired and checked by reading, not compiled, not run on hardware
//
#include <hip/hip_runtime.h>

// ---------------------------------------------------------------------------
// CDNA5 (gfx1250) APPNP pipeline, round 2.
//  - All GEMM B-operands stored K-major (B^T) -> no in-kernel transposition.
//  - Tiles staged to LDS with global_load_async_to_lds_b128 (ASYNCcnt),
//    double-buffered: next tile's async copies overlap current tile's WMMAs.
//  - v_wmma_f32_16x16x32_bf16, f32 accumulation, epilogue writes transposed
//    bf16 (vectorized, feeds next GEMM as B) and/or normal f32/bf16.
// ---------------------------------------------------------------------------

typedef __bf16 bf16;
typedef __attribute__((ext_vector_type(16))) bf16  v16bf;
typedef __attribute__((ext_vector_type(8)))  bf16  v8bf;
typedef __attribute__((ext_vector_type(4)))  bf16  v4bf;
typedef __attribute__((ext_vector_type(8)))  float v8f;

#define BM 128
#define BN 128
#define BK 32
#define LDSS 40   // bf16 elems per LDS row: 80B stride (16B aligned, bank-skewed)

// Async global->LDS copy of 16 bytes (tracked by ASYNCcnt, lands in LDS
// without touching VGPRs). lds operand is the 32-bit LDS byte offset.
__device__ __forceinline__ void async_copy_b128(const bf16* gaddr, const bf16* laddr) {
  unsigned lds = (unsigned)(size_t)(const void*)laddr;
  asm volatile("global_load_async_to_lds_b128 %0, %1, off"
               :: "v"(lds), "v"(gaddr) : "memory");
}
__device__ __forceinline__ void wait_async0() {
  asm volatile("s_wait_asynccnt 0x0" ::: "memory");
}

// ---------------------------------------------------------------------------
// f32 -> bf16 conversion (4-wide, grid-stride). n4 = n/4.
// ---------------------------------------------------------------------------
__global__ __launch_bounds__(256) void cvt_f32_bf16(
    const float* __restrict__ src, bf16* __restrict__ dst, long long n4)
{
  long long i = (long long)blockIdx.x * blockDim.x + threadIdx.x;
  long long stride = (long long)gridDim.x * blockDim.x;
  for (; i < n4; i += stride) {
    float4 v = *(const float4*)(src + 4 * i);
    v4bf o;
    o[0] = (bf16)v.x; o[1] = (bf16)v.y; o[2] = (bf16)v.z; o[3] = (bf16)v.w;
    *(v4bf*)(dst + 4 * i) = o;
  }
}

// f32 [R,C] row-major  ->  bf16 [C,R] (transposed). Small weight matrices only.
__global__ __launch_bounds__(256) void cvt_f32_bf16_T(
    const float* __restrict__ src, bf16* __restrict__ dst, int R, int C)
{
  int idx = blockIdx.x * blockDim.x + threadIdx.x;
  if (idx < R * C) {
    int r = idx / C, c = idx - r * C;
    dst[(size_t)c * R + r] = (bf16)src[idx];
  }
}

// ---------------------------------------------------------------------------
// C[M,N] = epilogue( A[M,K] @ Bt[N,K]^T ), bf16 inputs, f32 accum.
// Block tile 128x128x32; 256 threads = 8 waves; wave tile 32x64 (2x4 WMMA).
// Outputs: CbfT [N,M] (vector stores, feeds next GEMM as Bt),
//          Cbf  [M,N], Cf [M,N]. Requires M,N %128==0, K%32==0.
// ---------------------------------------------------------------------------
__global__ __launch_bounds__(256) void gemm_bf16_wmma(
    const bf16* __restrict__ A, const bf16* __restrict__ Bt,
    int M, int N, int K,
    const float* __restrict__ bias,     // [N] or nullptr
    const float* __restrict__ Z0,       // [M,N] f32 or nullptr
    float beta, float alpha,            // v = beta*v + alpha*Z0 when Z0 set
    int relu,
    bf16*  __restrict__ CbfT,           // [N,M] bf16 or nullptr
    bf16*  __restrict__ Cbf,            // [M,N] bf16 or nullptr
    float* __restrict__ Cf)             // [M,N] f32 or nullptr
{
  __shared__ bf16 sA[2][BM * LDSS];     // [m][k]
  __shared__ bf16 sB[2][BN * LDSS];     // [n][k]  (Bt rows are K-contiguous)

  const int tid  = threadIdx.x;
  const int lane = tid & 31;
  const int wave = tid >> 5;
  const int wm   = wave >> 1;           // 0..3 : 32-row slab
  const int wn   = wave & 1;            // 0..1 : 64-col slab
  const int m0   = blockIdx.y * BM;
  const int n0   = blockIdx.x * BN;

  // Issue the async copies for K-tile kt into LDS buffer `buf`.
  // 512 x b128 per matrix per tile -> 2 A-copies + 2 B-copies per thread.
  auto issue = [&](int kt, int buf) {
    const int k0 = kt * BK;
#pragma unroll
    for (int i = 0; i < 2; ++i) {
      int idx = tid + (i << 8);            // 0..511
      int r = idx >> 2, c = (idx & 3) << 3;
      async_copy_b128(A  + (size_t)(m0 + r) * K + (k0 + c), &sA[buf][r * LDSS + c]);
      async_copy_b128(Bt + (size_t)(n0 + r) * K + (k0 + c), &sB[buf][r * LDSS + c]);
    }
  };

  v8f acc[2][4];
#pragma unroll
  for (int mt = 0; mt < 2; ++mt)
#pragma unroll
    for (int nt = 0; nt < 4; ++nt)
#pragma unroll
      for (int i = 0; i < 8; ++i) acc[mt][nt][i] = 0.0f;

  const int KT = K / BK;
  issue(0, 0);
  wait_async0();
  __syncthreads();

  // Per-lane fragment addressing (CDNA5 wave32 WMMA layouts):
  //  A 16x32: lanes 0-15: row=lane, K={0..7,16..23}; lanes 16-31: K={8..15,24..31}
  //  B 32x16: lanes 0-15: col=lane, K={0..7,16..23}; lanes 16-31: K={8..15,24..31}
  const int r16 = lane & 15;
  const int akb = (lane < 16) ? 0 : 8;
  const int bkb = (lane < 16) ? 0 : 8;

  for (int kt = 0; kt < KT; ++kt) {
    const int cur = kt & 1;
    if (kt + 1 < KT) issue(kt + 1, cur ^ 1);   // overlaps WMMAs below

    v16bf afrag[2], bfrag[4];
#pragma unroll
    for (int mt = 0; mt < 2; ++mt) {
      const bf16* p = &sA[cur][(wm * 32 + mt * 16 + r16) * LDSS + akb];
      v8bf lo = *(const v8bf*)p;
      v8bf hi = *(const v8bf*)(p + 16);
#pragma unroll
      for (int j = 0; j < 8; ++j) { afrag[mt][j] = lo[j]; afrag[mt][j + 8] = hi[j]; }
    }
#pragma unroll
    for (int nt = 0; nt < 4; ++nt) {
      const bf16* p = &sB[cur][(wn * 64 + nt * 16 + r16) * LDSS + bkb];
      v8bf lo = *(const v8bf*)p;
      v8bf hi = *(const v8bf*)(p + 16);
#pragma unroll
      for (int j = 0; j < 8; ++j) { bfrag[nt][j] = lo[j]; bfrag[nt][j + 8] = hi[j]; }
    }

#pragma unroll
    for (int mt = 0; mt < 2; ++mt)
#pragma unroll
      for (int nt = 0; nt < 4; ++nt)
        acc[mt][nt] = __builtin_amdgcn_wmma_f32_16x16x32_bf16(
            false, afrag[mt], false, bfrag[nt], (short)0, acc[mt][nt],
            false, false);

    if (kt + 1 < KT) wait_async0();            // next tile landed in LDS
    __syncthreads();
  }

  // Epilogue. D layout: VGPR i, lanes 0-15 -> (M=i, N=lane),
  //                     lanes 16-31 -> (M=i+8, N=lane-16).
  // => per lane, acc[i] covers 8 consecutive M rows at one column: the
  //    transposed bf16 store is a single contiguous 16B vector store.
#pragma unroll
  for (int mt = 0; mt < 2; ++mt) {
#pragma unroll
    for (int nt = 0; nt < 4; ++nt) {
      const int gr0 = m0 + wm * 32 + mt * 16 + ((lane < 16) ? 0 : 8);
      const int gc  = n0 + wn * 64 + nt * 16 + r16;
      const float bv = bias ? bias[gc] : 0.0f;
      float v[8];
#pragma unroll
      for (int i = 0; i < 8; ++i) {
        float t = acc[mt][nt][i] + bv;
        if (Z0)   t = beta * t + alpha * Z0[(size_t)(gr0 + i) * N + gc];
        if (relu) t = fmaxf(t, 0.0f);
        v[i] = t;
      }
      if (CbfT) {
        v8bf o;
#pragma unroll
        for (int i = 0; i < 8; ++i) o[i] = (bf16)v[i];
        *(v8bf*)&CbfT[(size_t)gc * M + gr0] = o;
      }
      if (Cbf) {
#pragma unroll
        for (int i = 0; i < 8; ++i) Cbf[(size_t)(gr0 + i) * N + gc] = (bf16)v[i];
      }
      if (Cf) {
#pragma unroll
        for (int i = 0; i < 8; ++i) Cf[(size_t)(gr0 + i) * N + gc] = v[i];
      }
    }
  }
}

// ---------------------------------------------------------------------------
// Row-wise softmax over C=256 columns; one 256-thread block per row.
// ---------------------------------------------------------------------------
__global__ __launch_bounds__(256) void softmax_rows(
    const float* __restrict__ Z, float* __restrict__ out, int C)
{
  __shared__ float red[256];
  const int row = blockIdx.x;
  const int t   = threadIdx.x;
  const float v = Z[(size_t)row * C + t];

  red[t] = v; __syncthreads();
#pragma unroll
  for (int s = 128; s > 0; s >>= 1) {
    if (t < s) red[t] = fmaxf(red[t], red[t + s]);
    __syncthreads();
  }
  const float m = red[0];
  __syncthreads();

  const float e = __expf(v - m);
  red[t] = e; __syncthreads();
#pragma unroll
  for (int s = 128; s > 0; s >>= 1) {
    if (t < s) red[t] += red[t + s];
    __syncthreads();
  }
  out[(size_t)row * C + t] = e / red[0];
}

// ---------------------------------------------------------------------------
// Host orchestration (graph-capture safe: only kernel launches on `stream`).
// ---------------------------------------------------------------------------
extern "C" void kernel_launch(void* const* d_in, const int* in_sizes, int n_in,
                              void* d_out, int out_size, void* d_ws, size_t ws_size,
                              hipStream_t stream) {
  (void)in_sizes; (void)n_in; (void)out_size; (void)ws_size;

  const int N = 8192, F = 512, MLP = 512, C = 256;

  const float* features = (const float*)d_in[0];
  const float* fltr     = (const float*)d_in[1];
  const float* W0       = (const float*)d_in[2];
  const float* b0       = (const float*)d_in[3];
  const float* W1       = (const float*)d_in[4];
  const float* b1       = (const float*)d_in[5];
  const float* Wo       = (const float*)d_in[6];
  const float* bo       = (const float*)d_in[7];
  float* out = (float*)d_out;

  char* ws = (char*)d_ws;
  size_t off = 0;
  auto alloc = [&](size_t bytes) -> void* {
    void* p = ws + off;
    off += (bytes + 255) & ~(size_t)255;
    return p;
  };

  bf16*  A_bf  = (bf16*)alloc((size_t)N * N * 2);      // 128 MB, L2-resident
  bf16*  Xf_bf = (bf16*)alloc((size_t)N * F * 2);      // features (A-side)
  bf16*  W0t   = (bf16*)alloc((size_t)MLP * F * 2);    // W0^T [MLP,F]
  bf16*  W1t   = (bf16*)alloc((size_t)MLP * MLP * 2);  // W1^T
  bf16*  Wot   = (bf16*)alloc((size_t)C * MLP * 2);    // Wo^T [C,MLP]
  bf16*  Tt    = (bf16*)alloc((size_t)MLP * N * 2);    // (XW+b)^T [MLP,N]
  bf16*  X2    = (bf16*)alloc((size_t)N * MLP * 2);    // relu(A@T), A-side
  float* z0f   = (float*)alloc((size_t)N * C * 4);     // APPNP anchor [N,C]
  bf16*  zAt   = (bf16*)alloc((size_t)C * N * 2);      // z^T ping
  bf16*  zBt   = (bf16*)alloc((size_t)C * N * 2);      // z^T pong
  float* zf    = (float*)alloc((size_t)N * C * 4);     // final z, [N,C]

  // --- precision conversions (weights transposed to K-major) ---
  cvt_f32_bf16<<<8192, 256, 0, stream>>>(fltr,     A_bf,  (long long)N * N / 4);
  cvt_f32_bf16<<<2048, 256, 0, stream>>>(features, Xf_bf, (long long)N * F / 4);
  cvt_f32_bf16_T<<<(F * MLP + 255) / 256, 256, 0, stream>>>(W0, W0t, F, MLP);
  cvt_f32_bf16_T<<<(MLP * MLP + 255) / 256, 256, 0, stream>>>(W1, W1t, MLP, MLP);
  cvt_f32_bf16_T<<<(MLP * C + 255) / 256, 256, 0, stream>>>(Wo, Wot, MLP, C);

  const dim3 blk(256);
  const dim3 g512(MLP / BN, N / BM);   // (4, 64)
  const dim3 g256(C / BN,   N / BM);   // (2, 64)

  // --- MLP layer 0: T^T = (X@W0 + b0)^T ; X2 = relu(A @ T) ---
  gemm_bf16_wmma<<<g512, blk, 0, stream>>>(Xf_bf, W0t, N, MLP, F,
      b0, nullptr, 0.f, 0.f, 0, Tt, nullptr, nullptr);
  gemm_bf16_wmma<<<g512, blk, 0, stream>>>(A_bf, Tt, N, MLP, N,
      nullptr, nullptr, 0.f, 0.f, 1, nullptr, X2, nullptr);

  // --- MLP layer 1 ---
  gemm_bf16_wmma<<<g512, blk, 0, stream>>>(X2, W1t, N, MLP, MLP,
      b1, nullptr, 0.f, 0.f, 0, Tt, nullptr, nullptr);
  gemm_bf16_wmma<<<g512, blk, 0, stream>>>(A_bf, Tt, N, MLP, N,
      nullptr, nullptr, 0.f, 0.f, 1, nullptr, X2, nullptr);

  // --- output projection: z0 = X2@Wo + bo (f32 anchor + transposed bf16) ---
  gemm_bf16_wmma<<<g256, blk, 0, stream>>>(X2, Wot, N, C, MLP,
      bo, nullptr, 0.f, 0.f, 0, zAt, nullptr, z0f);

  // --- APPNP: z <- 0.9 * A@z + 0.1 * z0, 10 iterations (z kept transposed) ---
  bf16* zin = zAt;
  bf16* zot = zBt;
  for (int k = 0; k < 10; ++k) {
    float* cf = (k == 9) ? zf : nullptr;       // only final iterate -> softmax
    gemm_bf16_wmma<<<g256, blk, 0, stream>>>(A_bf, zin, N, C, N,
        nullptr, z0f, 0.9f, 0.1f, 0, zot, nullptr, cf);
    bf16* t = zin; zin = zot; zot = t;
  }

  // --- softmax over last dim ---
  softmax_rows<<<N, blk, 0, stream>>>(zf, out, C);
}
